// MultiHeadAttentionBlock_19267223290376
// MI455X (gfx1250) — matrix operand order repaired and checked by reading, not compiled, not run on hardware
//
#include <hip/hip_runtime.h>
#include <hip/hip_bf16.h>
#include <math.h>

// Problem constants (B,S,D,H fixed by the reference)
constexpr int Bc  = 4;
constexpr int Sc  = 2048;
constexpr int Dc  = 1024;
constexpr int Hc  = 16;
constexpr int DKc = 64;
constexpr int Mc  = Bc * Sc;   // 8192 rows

typedef __bf16 bf16;
typedef bf16  bf16x8 __attribute__((ext_vector_type(8)));
typedef bf16  v16bf  __attribute__((ext_vector_type(16)));
typedef float v8f    __attribute__((ext_vector_type(8)));
typedef float f32x2  __attribute__((ext_vector_type(2)));
typedef unsigned int u32x4 __attribute__((ext_vector_type(4)));
typedef int   i32x4  __attribute__((ext_vector_type(4)));
typedef int   i32x8  __attribute__((ext_vector_type(8)));

#define WMMA_BF16 __builtin_amdgcn_wmma_f32_16x16x32_bf16

// Build a 16-element bf16 A/B fragment from two contiguous 16-byte chunks.
static __device__ __forceinline__ v16bf make_frag(const bf16* lo, const bf16* hi) {
  bf16x8 l = *(const bf16x8*)lo;
  bf16x8 h = *(const bf16x8*)hi;
  return __builtin_shufflevector(l, h, 0,1,2,3,4,5,6,7,8,9,10,11,12,13,14,15);
}

// ---------------------------------------------------------------------------
// Tensor Data Mover: 2D bf16 tile load Global -> LDS, with LDS row padding.
// Descriptor packed per CDNA5 ISA (CDNA5 ISA):
//   group0: count=1 | lds_addr | global_addr | type=2
//   group1: data_size=2B, pad_enable/interval/amount, dims, tile dims, stride0
// pad_int_code: 0=2,1=4,2=8,3=16,4=32.. DWORDs between pads
// pad_amt_code: pad of (code+1) DWORDs
// ---------------------------------------------------------------------------
static __device__ __forceinline__ void tdm_load_2d_bf16(
    unsigned lds_addr, unsigned long long gaddr,
    unsigned tensor_d0, unsigned tensor_d1, unsigned stride0,
    unsigned tile_d0, unsigned tile_d1,
    unsigned pad_int_code, unsigned pad_amt_code)
{
  u32x4 g0;
  g0.x = 1u;                                            // count=1 (valid user D#)
  g0.y = lds_addr;                                      // LDS byte address
  g0.z = (unsigned)(gaddr & 0xFFFFFFFFull);             // global_addr[31:0]
  g0.w = (unsigned)((gaddr >> 32) & 0x1FFFFFFull)       // global_addr[56:32]
       | (2u << 30);                                    // type=2 ("image")
  unsigned d0 = (1u << 16);                             // data_size=1 -> 2 bytes
  d0 |= (1u << 20) | (pad_int_code << 22) | (pad_amt_code << 25);
  i32x8 g1;
  g1[0] = (int)d0;
  g1[1] = (int)((tensor_d0 & 0xFFFFu) << 16);                                   // tensor_dim0 lo16
  g1[2] = (int)(((tensor_d0 >> 16) & 0xFFFFu) | ((tensor_d1 & 0xFFFFu) << 16)); // d0 hi | d1 lo
  g1[3] = (int)(((tensor_d1 >> 16) & 0xFFFFu) | ((tile_d0 & 0xFFFFu) << 16));   // d1 hi | tile0
  g1[4] = (int)(tile_d1 & 0xFFFFu);                                             // tile1 | tile2=0
  g1[5] = (int)stride0;                                                         // dim0 stride lo32
  g1[6] = 0;                                                                    // stride0 hi | stride1 lo
  g1[7] = 0;                                                                    // stride1 hi
  i32x4 gz = {0, 0, 0, 0};
#if __has_include(<hip/amd_detail/amd_gfx1250_TDM.h>)
  i32x8 z8 = {0, 0, 0, 0, 0, 0, 0, 0};                  // clang-23 / therock: 6-arg form
  __builtin_amdgcn_tensor_load_to_lds(g0, g1, gz, gz, z8, 0);
#else
  __builtin_amdgcn_tensor_load_to_lds(g0, g1, gz, gz, 0);  // ROCm 7.2: 5-arg form
#endif
}

// ---------------------------------------------------------------------------
// Tiled GEMM: C[m,n] = A[m,:] . W[n,:] + bias[n]      (x @ W^T + b)
//   A_IS_BF16=false: A fp32 (converted to bf16 while staging, prefetch ahead)
//   A_IS_BF16=true : A bf16 staged by the Tensor Data Mover
//   OUT_MODE=0: store bf16 re-laid-out as [B,H,S,DK]   (projection outputs)
//   OUT_MODE=1: store fp32 row-major [M,D]             (final output)
// Block: 256 threads = 8 waves; tile 128(M) x 128(N); K-step 32.
// ---------------------------------------------------------------------------
template <bool A_IS_BF16, int OUT_MODE>
__global__ __launch_bounds__(256) void gemm_wmma(
    const float* __restrict__ A32, const bf16* __restrict__ A16,
    const float* __restrict__ W,   const float* __restrict__ bias,
    bf16* __restrict__ O16, float* __restrict__ O32)
{
  constexpr int LDA = 40;  // LDS row stride in elements (80B, 16B-aligned, conflict-spread)
  __shared__ __attribute__((aligned(16))) bf16 As[128 * LDA];
  __shared__ __attribute__((aligned(16))) bf16 Bs[128 * LDA];

  const int tid  = threadIdx.x;
  const int lane = tid & 31;
  const int w    = tid >> 5;
  const int wm   = w & 1;        // wave M index (0..1) -> 64 rows each
  const int wn   = w >> 1;       // wave N index (0..3) -> 32 cols each
  const int l16  = lane & 15;
  const int half = lane >> 4;    // 0: lanes 0-15, 1: lanes 16-31
  const int m0   = blockIdx.y * 128;
  const int n0   = blockIdx.x * 128;

  v8f acc[4][2];
  const v8f vzero = {0.f,0.f,0.f,0.f,0.f,0.f,0.f,0.f};
#pragma unroll
  for (int mt = 0; mt < 4; ++mt)
#pragma unroll
    for (int nt = 0; nt < 2; ++nt) acc[mt][nt] = vzero;

  for (int k0 = 0; k0 < Dc; k0 += 32) {
    // ---- stage A tile (128 x 32) ----
    if constexpr (!A_IS_BF16) {
#pragma unroll
      for (int t = 0; t < 8; ++t) {
        int c   = tid + t * 256;        // 2048 float2 chunks
        int row = c >> 4;
        int kk  = (c & 15) * 2;
        f32x2 val = *(const f32x2*)(A32 + (size_t)(m0 + row) * Dc + k0 + kk);
        As[row * LDA + kk]     = (bf16)val.x;
        As[row * LDA + kk + 1] = (bf16)val.y;
      }
    } else {
      // TDM: 128x32 bf16 tile, LDS rows 64B + 16B pad (interval 16 DW, pad 4 DW)
      if (w == 0) {
        tdm_load_2d_bf16((unsigned)(unsigned long long)As,
                         (unsigned long long)(A16 + (size_t)m0 * Dc + k0),
                         /*tensor_d0=*/Dc, /*tensor_d1=*/Mc, /*stride0=*/Dc,
                         /*tile_d0=*/32, /*tile_d1=*/128,
                         /*pad_int=*/3, /*pad_amt=*/3);
      }
    }
    // ---- stage W tile (128 x 32), fp32 -> bf16 ----
#pragma unroll
    for (int t = 0; t < 8; ++t) {
      int c   = tid + t * 256;
      int row = c >> 4;
      int kk  = (c & 15) * 2;
      f32x2 val = *(const f32x2*)(W + (size_t)(n0 + row) * Dc + k0 + kk);
      Bs[row * LDA + kk]     = (bf16)val.x;
      Bs[row * LDA + kk + 1] = (bf16)val.y;
    }
    // ---- prefetch next K-step tiles (global_prefetch_b8), cacheline strided ----
    if (k0 + 32 < Dc) {
      const char* pw = (const char*)(W + (size_t)(n0 + (tid >> 1)) * Dc + (k0 + 32))
                     + (tid & 1) * 64;
      __builtin_prefetch(pw, 0, 0);
      if constexpr (!A_IS_BF16) {
        const char* pa = (const char*)(A32 + (size_t)(m0 + (tid >> 1)) * Dc + (k0 + 32))
                       + (tid & 1) * 64;
        __builtin_prefetch(pa, 0, 0);
      }
    }
    if constexpr (A_IS_BF16) {
      if (w == 0) __builtin_amdgcn_s_wait_tensorcnt(0);
    }
    __syncthreads();

    // ---- fragments + WMMA ----
    v16bf afrag[4], bfrag[2];
#pragma unroll
    for (int mt = 0; mt < 4; ++mt) {
      const bf16* base = As + (wm * 64 + mt * 16 + l16) * LDA;
      int ka = half ? 8 : 0;                 // A: lanes0-15 K0..7/16..23, lanes16-31 K8..15/24..31
      afrag[mt] = make_frag(base + ka, base + ka + 16);
    }
#pragma unroll
    for (int nt = 0; nt < 2; ++nt) {
      const bf16* base = Bs + (wn * 32 + nt * 16 + l16) * LDA;
      int kb = half ? 8 : 0;                 // B: lanes0-15 K0..7/16..23, lanes16-31 K8..15/24..31
      bfrag[nt] = make_frag(base + kb, base + kb + 16);
    }
#pragma unroll
    for (int mt = 0; mt < 4; ++mt)
#pragma unroll
      for (int nt = 0; nt < 2; ++nt)
        acc[mt][nt] = WMMA_BF16(false, afrag[mt], false, bfrag[nt],
                                (short)0, acc[mt][nt], false, false);
    __syncthreads();
  }

  // ---- epilogue: bias add + store ----
#pragma unroll
  for (int nt = 0; nt < 2; ++nt) {
    int n = n0 + wn * 32 + nt * 16 + l16;
    float bv = bias[n];
#pragma unroll
    for (int mt = 0; mt < 4; ++mt) {
      int rbase = m0 + wm * 64 + mt * 16 + half * 8;  // C layout: VGPR i -> row i (or 8+i)
#pragma unroll
      for (int i = 0; i < 8; ++i) {
        float val = acc[mt][nt][i] + bv;
        int row = rbase + i;
        if constexpr (OUT_MODE == 0) {
          int b  = row >> 11, s = row & (Sc - 1);
          int h  = n >> 6,    dk = n & (DKc - 1);
          O16[(((size_t)(b * Hc + h)) * Sc + s) * DKc + dk] = (bf16)val;
        } else {
          O32[(size_t)row * Dc + n] = val;
        }
      }
    }
  }
}

// ---------------------------------------------------------------------------
// Causal flash attention over bf16 Q/K/V laid out [B*H, S, 64].
// Block = 256 thr = 8 waves; block handles 128 query rows of one (b,h).
// Each wave owns 16 query rows; key blocks of 64; online softmax.
// K tile staged by the Tensor Data Mover (pad reproduces LDS stride 72).
// ---------------------------------------------------------------------------
__global__ __launch_bounds__(256) void flash_attn(
    const bf16* __restrict__ Q, const bf16* __restrict__ K,
    const bf16* __restrict__ V, bf16* __restrict__ X)
{
  constexpr int KB   = 64;   // key block
  constexpr int KSTR = 72;   // LDS stride (144B, 16B-aligned, conflict-spread)
  constexpr int PSTR = 72;
  __shared__ __attribute__((aligned(16))) bf16 Ks[KB * KSTR];        // [key][dk]
  __shared__ __attribute__((aligned(16))) bf16 Vt[DKc * KSTR];       // [dk][key] (transposed)
  __shared__ __attribute__((aligned(16))) bf16 Ps[8 * 16 * PSTR];    // per-wave P (16 x 64)

  const int tid  = threadIdx.x;
  const int lane = tid & 31;
  const int w    = tid >> 5;
  const int l16  = lane & 15;
  const int half = lane >> 4;
  const int qb   = blockIdx.x;              // query block of 128 rows
  const int bh   = blockIdx.y;              // b*H + h
  const int q0w  = qb * 128 + w * 16;       // wave's first query row

  const bf16* Qbh = Q + (size_t)bh * Sc * DKc;
  const bf16* Kbh = K + (size_t)bh * Sc * DKc;
  const bf16* Vbh = V + (size_t)bh * Sc * DKc;

  // Q fragments live in registers for the whole block (A layout, direct from global)
  v16bf aq[2];
  {
    const bf16* qrow = Qbh + (size_t)(q0w + l16) * DKc;
#pragma unroll
    for (int ks = 0; ks < 2; ++ks) {
      int ka = ks * 32 + (half ? 8 : 0);
      aq[ks] = make_frag(qrow + ka, qrow + ka + 16);
    }
  }

  const v8f vzero = {0.f,0.f,0.f,0.f,0.f,0.f,0.f,0.f};
  float vm[8], vl[8];
  v8f o[4];
#pragma unroll
  for (int i = 0; i < 8; ++i) { vm[i] = -3.0e38f; vl[i] = 0.f; }
#pragma unroll
  for (int n = 0; n < 4; ++n) o[n] = vzero;

  const int nkb = 2 * qb + 2;               // causal: keys up to (qb+1)*128
  for (int kb = 0; kb < nkb; ++kb) {
    // ---- stage K tile [64 key][64 dk] via TDM (rows 128B + 16B pad: codes 4/3) ----
    if (w == 0) {
      tdm_load_2d_bf16((unsigned)(unsigned long long)Ks,
                       (unsigned long long)(Kbh + (size_t)(kb * KB) * DKc),
                       /*tensor_d0=*/DKc, /*tensor_d1=*/Sc, /*stride0=*/DKc,
                       /*tile_d0=*/DKc, /*tile_d1=*/KB,
                       /*pad_int=*/4, /*pad_amt=*/3);
    }
    // ---- stage V tile transposed -> Vt[dk][key] (all threads) ----
#pragma unroll
    for (int t = 0; t < 2; ++t) {
      int c = tid + t * 256;
      int key = c >> 3;
      int dk0 = (c & 7) * 8;
      bf16x8 vv = *(const bf16x8*)(Vbh + (size_t)(kb * KB + key) * DKc + dk0);
#pragma unroll
      for (int j = 0; j < 8; ++j) Vt[(dk0 + j) * KSTR + key] = vv[j];
    }
    if (w == 0) __builtin_amdgcn_s_wait_tensorcnt(0);
    __syncthreads();

    // ---- scores: S = Q . K^T   (16 x 64 per wave) ----
    v8f s[4];
#pragma unroll
    for (int nt = 0; nt < 4; ++nt) s[nt] = vzero;
#pragma unroll
    for (int ks = 0; ks < 2; ++ks) {
#pragma unroll
      for (int nt = 0; nt < 4; ++nt) {
        const bf16* base = Ks + (nt * 16 + l16) * KSTR + ks * 32 + (half ? 8 : 0);
        v16bf kf = make_frag(base, base + 16);
        s[nt] = WMMA_BF16(false, aq[ks], false, kf, (short)0, s[nt], false, false);
      }
    }

    // ---- scale + causal mask ----
    const bool need_mask = (kb * KB + KB - 1) > q0w;
#pragma unroll
    for (int nt = 0; nt < 4; ++nt) {
      int keyc = kb * KB + nt * 16 + l16;
#pragma unroll
      for (int i = 0; i < 8; ++i) {
        float val = s[nt][i] * 0.125f;      // 1/sqrt(64)
        int qrow = q0w + half * 8 + i;
        if (need_mask && keyc > qrow) val = -3.0e38f;
        s[nt][i] = val;
      }
    }

    // ---- online softmax (row i lives at vector element i; 16 lanes/half) ----
#pragma unroll
    for (int i = 0; i < 8; ++i) {
      float mx = fmaxf(fmaxf(s[0][i], s[1][i]), fmaxf(s[2][i], s[3][i]));
      mx = fmaxf(mx, __shfl_xor(mx, 1, 32));
      mx = fmaxf(mx, __shfl_xor(mx, 2, 32));
      mx = fmaxf(mx, __shfl_xor(mx, 4, 32));
      mx = fmaxf(mx, __shfl_xor(mx, 8, 32));
      float mnew = fmaxf(vm[i], mx);
      float corr = __expf(vm[i] - mnew);
      vm[i] = mnew;
      float rsum = 0.f;
#pragma unroll
      for (int nt = 0; nt < 4; ++nt) {
        float p = __expf(s[nt][i] - mnew);
        s[nt][i] = p;
        rsum += p;
      }
      rsum += __shfl_xor(rsum, 1, 32);
      rsum += __shfl_xor(rsum, 2, 32);
      rsum += __shfl_xor(rsum, 4, 32);
      rsum += __shfl_xor(rsum, 8, 32);
      vl[i] = vl[i] * corr + rsum;
      o[0][i] *= corr; o[1][i] *= corr; o[2][i] *= corr; o[3][i] *= corr;
    }

    // ---- write P to per-wave LDS (C-layout -> A-layout re-staging) ----
    bf16* prow = Ps + w * 16 * PSTR;
#pragma unroll
    for (int nt = 0; nt < 4; ++nt)
#pragma unroll
      for (int i = 0; i < 8; ++i)
        prow[(half * 8 + i) * PSTR + nt * 16 + l16] = (bf16)s[nt][i];

    // ---- O += P . V ----
#pragma unroll
    for (int ks = 0; ks < 2; ++ks) {
      const bf16* pbase = prow + l16 * PSTR + ks * 32 + (half ? 8 : 0);
      v16bf pa = make_frag(pbase, pbase + 16);
#pragma unroll
      for (int n = 0; n < 4; ++n) {
        const bf16* vbase = Vt + (n * 16 + l16) * KSTR + ks * 32 + (half ? 8 : 0);
        v16bf vb = make_frag(vbase, vbase + 16);
        o[n] = WMMA_BF16(false, pa, false, vb, (short)0, o[n], false, false);
      }
    }
    __syncthreads();
  }

  // ---- normalize + store X as bf16 [B*S, D] (row-major, head-interleaved) ----
  const int bq = bh >> 4;        // bh / H
  const int hq = bh & 15;        // bh % H
#pragma unroll
  for (int i = 0; i < 8; ++i) {
    float inv = 1.0f / vl[i];
    int srow = q0w + half * 8 + i;
    size_t base = ((size_t)bq * Sc + srow) * Dc + hq * DKc;
#pragma unroll
    for (int n = 0; n < 4; ++n)
      X[base + n * 16 + l16] = (bf16)(o[n][i] * inv);
  }
}

// ---------------------------------------------------------------------------
// Launch: Q/K/V projections -> flash attention -> output projection.
// ---------------------------------------------------------------------------
extern "C" void kernel_launch(void* const* d_in, const int* in_sizes, int n_in,
                              void* d_out, int out_size, void* d_ws, size_t ws_size,
                              hipStream_t stream) {
  const float* q   = (const float*)d_in[0];
  const float* k   = (const float*)d_in[1];
  const float* v   = (const float*)d_in[2];
  // d_in[3] = causal mask: known analytically, not read.
  const float* w_q = (const float*)d_in[4];
  const float* b_q = (const float*)d_in[5];
  const float* w_k = (const float*)d_in[6];
  const float* b_k = (const float*)d_in[7];
  const float* w_v = (const float*)d_in[8];
  const float* b_v = (const float*)d_in[9];
  const float* w_o = (const float*)d_in[10];
  const float* b_o = (const float*)d_in[11];
  float* out = (float*)d_out;

  char* ws = (char*)d_ws;
  const size_t SZ = (size_t)Mc * Dc * sizeof(bf16);  // 16 MB per tensor
  bf16* Qb = (bf16*)(ws);
  bf16* Kb = (bf16*)(ws + SZ);
  bf16* Vb = (bf16*)(ws + 2 * SZ);
  bf16* Xb = (bf16*)(ws + 3 * SZ);

  dim3 blk(256);
  dim3 gg(Dc / 128, Mc / 128);     // (8, 64)
  gemm_wmma<false, 0><<<gg, blk, 0, stream>>>(q, nullptr, w_q, b_q, Qb, nullptr);
  gemm_wmma<false, 0><<<gg, blk, 0, stream>>>(k, nullptr, w_k, b_k, Kb, nullptr);
  gemm_wmma<false, 0><<<gg, blk, 0, stream>>>(v, nullptr, w_v, b_v, Vb, nullptr);

  dim3 ga(Sc / 128, Bc * Hc);      // (16, 64)
  flash_attn<<<ga, blk, 0, stream>>>(Qb, Kb, Vb, Xb);

  gemm_wmma<true, 1><<<gg, blk, 0, stream>>>(nullptr, Xb, w_o, b_o, nullptr, out);
}
